// MambaBlock_84576495993368
// MI455X (gfx1250) — matrix operand order repaired and checked by reading, not compiled, not run on hardware
//
#include <hip/hip_runtime.h>

// ---------------- problem constants (B=2, L=2048, D=2048, N=16, K=4) --------------------
#define B_SZ   2
#define L_SEQ  2048
#define D_DIM  2048
#define N_DIM  16
#define DOUT   (2 * D_DIM + N_DIM + 1)   // 4113
#define M_TOT  (B_SZ * L_SEQ)            // 4096
#define NPAD   4224                      // 33 * 128, padded GEMM-N / xp row stride

// ---------------- GEMM tiling -----------------------------------------------------------
#define BM   128
#define BN   128
#define KT   64    // two 32-deep WMMA steps per LDS stage
#define LDSW 72    // padded LDS stride in halfwords: 144B rows, 16B-aligned b128 frags

typedef __attribute__((ext_vector_type(16))) __bf16 v16bf;
typedef __attribute__((ext_vector_type(8)))  float  v8f;
typedef __attribute__((ext_vector_type(4)))  int    v4i;

// fp32 -> bf16 round-to-nearest-even
__device__ __forceinline__ unsigned short f2bf(float f) {
    unsigned int u = __float_as_uint(f);
    unsigned int r = u + 0x7FFFu + ((u >> 16) & 1u);
    return (unsigned short)(r >> 16);
}

// ---------------- CDNA5 async global->LDS staging (fallback: vector copy) ---------------
#if __has_builtin(__builtin_amdgcn_global_load_async_to_lds_b128)
#define USE_ASYNC 1
#else
#define USE_ASYNC 0
#endif

__device__ __forceinline__ void stage16(const unsigned short* g, unsigned short* l) {
#if USE_ASYNC
    __builtin_amdgcn_global_load_async_to_lds_b128(
        (__attribute__((address_space(1))) v4i*)(void*)g,
        (__attribute__((address_space(3))) v4i*)(void*)l, 0, 0);
#else
    *(uint4*)l = *(const uint4*)g;
#endif
}

// wait until at most 8 async ops outstanding (one full tile still in flight)
__device__ __forceinline__ void wait_async_le8() {
#if USE_ASYNC
#if __has_builtin(__builtin_amdgcn_s_wait_asynccnt)
    __builtin_amdgcn_s_wait_asynccnt(8);
#else
    asm volatile("s_wait_asynccnt 8" ::: "memory");
#endif
#endif
}

__device__ __forceinline__ void wait_async_le0() {
#if USE_ASYNC
#if __has_builtin(__builtin_amdgcn_s_wait_asynccnt)
    __builtin_amdgcn_s_wait_asynccnt(0);
#else
    asm volatile("s_wait_asynccnt 0" ::: "memory");
#endif
#endif
}

// A fragment, 16x32 bf16 (CDNA5 ISA "16-bit A-Matrix 16x32"):
// lanes 0-15: row M=l, v0-3 = K 0..7 (pairs), v4-7 = K 16..23 ; lanes 16-31: K 8..15 / 24..31
__device__ __forceinline__ v16bf load_frag_A(const unsigned short* rowbase, int hi) {
    union { v16bf v; unsigned int u[8]; } r;
    const unsigned int* p = (const unsigned int*)rowbase;
    const int o1 = hi * 4, o2 = 8 + hi * 4;
#pragma unroll
    for (int j = 0; j < 4; ++j) { r.u[j] = p[o1 + j]; r.u[4 + j] = p[o2 + j]; }
    return r.v;
}

// B fragment, 32x16 bf16: lanes 0-15: col N=l, K 0..7 / 16..23 ; lanes 16-31: col N=l, K 8..15 / 24..31.
// sB is [n][k] so each lane's K-halfwords are contiguous.
__device__ __forceinline__ v16bf load_frag_B(const unsigned short* colbase, int hi) {
    union { v16bf v; unsigned int u[8]; } r;
    const unsigned int* p = (const unsigned int*)colbase;
    const int o1 = hi * 4, o2 = 8 + hi * 4;
#pragma unroll
    for (int j = 0; j < 4; ++j) { r.u[j] = p[o1 + j]; r.u[4 + j] = p[o2 + j]; }
    return r.v;
}

// ============================ Kernel 0a: X fp32 -> bf16 =================================
__global__ __launch_bounds__(256)
void convert_x_bf16(const float* __restrict__ X, unsigned short* __restrict__ Xbf) {
    const size_t i = ((size_t)blockIdx.x * 256 + threadIdx.x) * 4;
    const float4 f = *(const float4*)(X + i);
    union { unsigned short h[4]; uint2 v; } o;
    o.h[0] = f2bf(f.x); o.h[1] = f2bf(f.y); o.h[2] = f2bf(f.z); o.h[3] = f2bf(f.w);
    *(uint2*)(Xbf + i) = o.v;
}

// ============================ Kernel 0b: W fp32 (K,N) -> bf16 (NPAD,K) ==================
__global__ __launch_bounds__(256)
void convert_transpose_w(const float* __restrict__ W, unsigned short* __restrict__ Wt) {
    __shared__ unsigned short t[32][33];
    const int k0 = blockIdx.x * 32, n0 = blockIdx.y * 32;
    const int tx = threadIdx.x, ty = threadIdx.y;      // (32, 8)
#pragma unroll
    for (int i = 0; i < 4; ++i) {
        const int n = n0 + tx;
        const float f = (n < DOUT) ? W[(size_t)(k0 + ty + i * 8) * DOUT + n] : 0.0f;
        t[ty + i * 8][tx] = f2bf(f);
    }
    __syncthreads();
#pragma unroll
    for (int i = 0; i < 4; ++i)
        Wt[(size_t)(n0 + ty + i * 8) * D_DIM + k0 + tx] = t[tx][ty + i * 8];
}

// ============================ Kernel 1: xp = x @ W_in + b_in (bf16 WMMA) ================
__global__ __launch_bounds__(256)
void mamba_inproj_wmma(const unsigned short* __restrict__ Xbf, // (4096, 2048) bf16
                       const unsigned short* __restrict__ Wt,  // (NPAD, 2048) bf16 col-major W
                       const float* __restrict__ bias,         // (4113)
                       float* __restrict__ XP)                 // (4096, NPAD)
{
    __shared__ alignas(16) unsigned short sA[2][BM * LDSW];   // double-buffered
    __shared__ alignas(16) unsigned short sB[2][BN * LDSW];

    const int m0   = blockIdx.x * BM;
    const int n0   = blockIdx.y * BN;
    const int tid  = threadIdx.x;
    const int lane = tid & 31;
    const int wid  = tid >> 5;
    const int wm   = (wid & 1) * 64;        // wave sub-tile 64 x 32
    const int wn   = (wid >> 1) * 32;
    const int l15  = lane & 15;
    const int hi   = lane >> 4;

    // per-thread staging coordinates (8 x 16B chunks per tile: 4 for A, 4 for B)
    v8f acc[4][2] = {};

    // ---- issue tile 0 ------------------------------------------------------------------
#pragma unroll
    for (int i = 0; i < 4; ++i) {
        const int u = tid + i * 256, r = u >> 3, seg = (u & 7) * 8;
        stage16(Xbf + (size_t)(m0 + r) * D_DIM + seg, sA[0] + r * LDSW + seg);
    }
#pragma unroll
    for (int i = 0; i < 4; ++i) {
        const int u = tid + i * 256, r = u >> 3, seg = (u & 7) * 8;
        stage16(Wt + (size_t)(n0 + r) * D_DIM + seg, sB[0] + r * LDSW + seg);
    }

    int buf = 0;
    for (int k0 = 0; k0 < D_DIM; k0 += KT) {
        const bool more = (k0 + KT) < D_DIM;
        // ---- issue tile k0+KT into the other buffer (overlaps with compute below) -----
        if (more) {
            const int kn = k0 + KT;
#pragma unroll
            for (int i = 0; i < 4; ++i) {
                const int u = tid + i * 256, r = u >> 3, seg = (u & 7) * 8;
                stage16(Xbf + (size_t)(m0 + r) * D_DIM + kn + seg, sA[buf ^ 1] + r * LDSW + seg);
            }
#pragma unroll
            for (int i = 0; i < 4; ++i) {
                const int u = tid + i * 256, r = u >> 3, seg = (u & 7) * 8;
                stage16(Wt + (size_t)(n0 + r) * D_DIM + kn + seg, sB[buf ^ 1] + r * LDSW + seg);
            }
            wait_async_le8();   // in-order completion => current tile has landed
        } else {
            wait_async_le0();
        }
        __syncthreads();

        // ---- 16 WMMAs per wave on the current buffer ----------------------------------
        const unsigned short* cA = sA[buf];
        const unsigned short* cB = sB[buf];
#pragma unroll
        for (int ks = 0; ks < 2; ++ks) {
            const int ko = ks * 32;
            v16bf bf[2];
#pragma unroll
            for (int j = 0; j < 2; ++j)
                bf[j] = load_frag_B(&cB[(wn + j * 16 + l15) * LDSW + ko], hi);
#pragma unroll
            for (int i = 0; i < 4; ++i) {
                v16bf af = load_frag_A(&cA[(wm + i * 16 + l15) * LDSW + ko], hi);
#pragma unroll
                for (int j = 0; j < 2; ++j)
                    acc[i][j] = __builtin_amdgcn_wmma_f32_16x16x32_bf16(
                        false, af, false, bf[j], (short)0, acc[i][j], false, false);
            }
        }
        __syncthreads();        // buffer safe to overwrite next iteration
        buf ^= 1;
    }

    // ---- store C (+bias) into padded XP; C layout: VGPR g -> row g + 8*hi, col l15 ----
#pragma unroll
    for (int i = 0; i < 4; ++i) {
#pragma unroll
        for (int j = 0; j < 2; ++j) {
            const int col = n0 + wn + j * 16 + l15;
            const float bv = (col < DOUT) ? bias[col] : 0.0f;
            const int rbase = m0 + wm + i * 16 + hi * 8;
#pragma unroll
            for (int g = 0; g < 8; ++g)
                XP[(size_t)(rbase + g) * NPAD + col] = acc[i][j][g] + bv;
        }
    }
}

// ============================ Kernel 2: silu/conv/selective-scan ========================
__global__ __launch_bounds__(256)
void mamba_scan(const float* __restrict__ XP,    // (4096, NPAD)
                const float* __restrict__ Alog, const float* __restrict__ Bp,
                const float* __restrict__ Cp,   const float* __restrict__ CW,
                float* __restrict__ Y)           // (B, L, D)
{
    const int idx = blockIdx.x * blockDim.x + threadIdx.x;   // over B*D*N = 65536
    const int n = idx & (N_DIM - 1);
    const int d = (idx >> 4) & (D_DIM - 1);
    const int b = idx >> 15;

    const float Ac = -__expf(Alog[d * N_DIM + n]);
    const float Cc = Cp[d * N_DIM + n];
    const float Bn = Bp[n];
    const float w0 = CW[d * 4 + 0], w1 = CW[d * 4 + 1],
                w2 = CW[d * 4 + 2], w3 = CW[d * 4 + 3];

    float x1 = 0.f, x2 = 0.f, x3 = 0.f;
    float h = 0.f;
    const float* base = XP + (size_t)b * L_SEQ * NPAD;

    for (int t = 0; t < L_SEQ; ++t) {
        const float* row = base + (size_t)t * NPAD;

        const float draw  = row[2 * D_DIM + N_DIM];
        const float sp    = (draw > 20.f) ? draw : __logf(1.f + __expf(draw));
        const float delta = sp * (1.f / (1.f + __expf(-sp)));

        const float bt = row[2 * D_DIM + n] + Bn;

        const float xc = row[D_DIM + d];
        const float sx = xc * (1.f / (1.f + __expf(-xc)));
        const float xv = x1 * w0 + x2 * w1 + x3 * w2 + sx * w3;
        x1 = x2; x2 = x3; x3 = sx;

        const float dA = __expf(delta * Ac);
        h = dA * h + (delta * bt) * xv;
        float yp = h * Cc;
        yp += __shfl_xor(yp, 1);
        yp += __shfl_xor(yp, 2);
        yp += __shfl_xor(yp, 4);
        yp += __shfl_xor(yp, 8);

        if (n == 0) {
            const float v  = row[d];
            const float sv = v * (1.f / (1.f + __expf(-v)));
            Y[((size_t)b * L_SEQ + t) * D_DIM + d] = yp * sv;
        }
    }
}

// ============================ launch ====================================================
extern "C" void kernel_launch(void* const* d_in, const int* in_sizes, int n_in,
                              void* d_out, int out_size, void* d_ws, size_t ws_size,
                              hipStream_t stream) {
    const float* x    = (const float*)d_in[0];
    const float* W    = (const float*)d_in[1];
    const float* bi   = (const float*)d_in[2];
    const float* Alog = (const float*)d_in[3];
    const float* Bp   = (const float*)d_in[4];
    const float* Cp   = (const float*)d_in[5];
    const float* cw   = (const float*)d_in[6];
    float* out = (float*)d_out;

    // workspace layout: xp (padded f32) | Xbf | Wt   (~103 MB total)
    float*          xp  = (float*)d_ws;
    unsigned short* Xbf = (unsigned short*)((char*)d_ws + (size_t)M_TOT * NPAD * 4);
    unsigned short* Wt  = Xbf + (size_t)M_TOT * D_DIM;

    // 0a: x -> bf16 (8.4M elems, 4/thread)
    convert_x_bf16<<<(M_TOT * (size_t)D_DIM) / 4 / 256, 256, 0, stream>>>(x, Xbf);
    // 0b: W -> bf16 transposed, padded (grid 64 x 132 of 32x32 tiles)
    convert_transpose_w<<<dim3(D_DIM / 32, NPAD / 32), dim3(32, 8), 0, stream>>>(W, Wt);
    // 1: GEMM (double-buffered async LDS pipeline)
    mamba_inproj_wmma<<<dim3(M_TOT / BM, NPAD / BN), 256, 0, stream>>>(Xbf, Wt, bi, xp);
    // 2: scan
    mamba_scan<<<(B_SZ * D_DIM * N_DIM) / 256, 256, 0, stream>>>(xp, Alog, Bp, Cp, cw, out);
}
